// EdgeGNN_49409303773978
// MI455X (gfx1250) — matrix operand order repaired and checked by reading, not compiled, not run on hardware
//
#include <hip/hip_runtime.h>
#include <hip/hip_bf16.h>

typedef __attribute__((ext_vector_type(16))) __bf16 v16bf;
typedef __attribute__((ext_vector_type(8)))  __bf16 v8bf;
typedef __attribute__((ext_vector_type(8)))  float  v8f;
typedef unsigned int uint32;
typedef __attribute__((ext_vector_type(4))) uint32 u32x4;
typedef __attribute__((ext_vector_type(8))) uint32 u32x8;

#define N_NODES 50000
#define N_EDGES 800000
#define IN_NODE 16
#define IN_EDGE 8
#define HID     128
#define N_LAYERS 4
#define LN_EPS  1e-5f
#define K1PAD   288     // 2*HID + IN_EDGE = 264, padded to 9*32

// ---------------- TDM: 1-D bf16 tile (n2 elements of 2 bytes) global -> LDS ----------------
static __device__ __forceinline__ void tdm_load_bf16_1d(uint32 lds_off, const void* gptr, uint32 n2) {
    unsigned long long ga = (unsigned long long)(uintptr_t)gptr;
    u32x4 g0;
    g0[0] = 1u;                                              // count=1 (valid user descriptor)
    g0[1] = lds_off;                                         // lds_addr (bytes)
    g0[2] = (uint32)(ga & 0xFFFFFFFFu);                      // global_addr[31:0]
    g0[3] = (uint32)((ga >> 32) & 0x01FFFFFFu) | (2u << 30); // global_addr[56:32] | type=2
    u32x8 g1;
    g1[0] = 1u << 16;                                        // workgroup_mask=0, data_size=1 (2 bytes)
    g1[1] = (n2 & 0xFFFFu) << 16;                            // tensor_dim0[15:0]
    g1[2] = ((n2 >> 16) & 0xFFFFu) | (1u << 16);             // tensor_dim0[31:16], tensor_dim1=1
    g1[3] = (n2 & 0xFFFFu) << 16;                            // tile_dim0 = n2
    g1[4] = 0u;                                              // tile_dim1/2 unused
    g1[5] = n2;                                              // tensor_dim0_stride[31:0]
    g1[6] = 0u;
    g1[7] = 0u;
    asm volatile("tensor_load_to_lds %0, %1" :: "s"(g0), "s"(g1) : "memory");
}
static __device__ __forceinline__ uint32 lds_offset_of(const void* p) {
    return (uint32)(uintptr_t)p;   // LDS generic pointers: low 32 bits = LDS byte offset
}

// ---------------- WMMA operand loaders (CDNA5 ISA 7.12.2, wave32) ----------------
// A 16x32 bf16: lane=(row=lane&15, kg=lane>>4); elems j<8 -> K=kb+kg*8+j ; j>=8 -> K=kb+16+kg*8+(j-8)
static __device__ __forceinline__ v16bf loadA_bf16(const __bf16* __restrict__ rowp, int kb, int kg) {
    v8bf lo = *(const v8bf*)(rowp + kb + kg * 8);
    v8bf hi = *(const v8bf*)(rowp + kb + 16 + kg * 8);
    v16bf a;
#pragma unroll
    for (int j = 0; j < 8; ++j) { a[j] = lo[j]; a[8 + j] = hi[j]; }
    return a;
}
// B 32x16 bf16, B[k][n] = Wrow_n[k]: lane=(n=lane&15, kg=lane>>4); elems j<8 -> K=kb+kg*8+j ; j>=8 -> K=kb+16+kg*8+(j-8)
static __device__ __forceinline__ v16bf loadB_bf16(const __bf16* __restrict__ wrow, int kb, int kg) {
    v8bf lo = *(const v8bf*)(wrow + kb + kg * 8);
    v8bf hi = *(const v8bf*)(wrow + kb + 16 + kg * 8);
    v16bf b;
#pragma unroll
    for (int j = 0; j < 8; ++j) { b[j] = lo[j]; b[8 + j] = hi[j]; }
    return b;
}

// ---------------- weight conversion kernels (run once per launch) ----------------
__global__ void EdgeGNN_cvt_kernel(const float* __restrict__ src, __bf16* __restrict__ dst, int n) {
    int i = blockIdx.x * 256 + threadIdx.x;
    if (i < n) dst[i] = (__bf16)src[i];
}
__global__ void EdgeGNN_cvtpad_w1_kernel(const float* __restrict__ W1, __bf16* __restrict__ W1p) {
    int i = blockIdx.x * 256 + threadIdx.x;            // HID * K1PAD
    if (i < HID * K1PAD) {
        int n = i / K1PAD, k = i % K1PAD;
        W1p[i] = (k < 264) ? (__bf16)W1[n * 264 + k] : (__bf16)0.f;
    }
}

// ---------------- prep kernels ----------------
__global__ void EdgeGNN_deg_kernel(const int* __restrict__ dstI, float* __restrict__ deg) {
    int e = blockIdx.x * 256 + threadIdx.x;
    if (e < N_EDGES) atomicAdd(&deg[dstI[e]], 1.0f);
}
__global__ void EdgeGNN_invdeg_kernel(float* __restrict__ deg) {
    int n = blockIdx.x * 256 + threadIdx.x;
    if (n < N_NODES) deg[n] = 1.0f / fmaxf(deg[n], 1.0f);
}
__global__ void EdgeGNN_inproj_kernel(const float* __restrict__ x, const float* __restrict__ Win,
                                      const float* __restrict__ bin, __bf16* __restrict__ h) {
    const int n = blockIdx.x;
    const int t = threadIdx.x;                         // 0..127
    const float* xr = x + n * IN_NODE;
    const float* wr = Win + t * IN_NODE;
    float acc = bin[t];
#pragma unroll
    for (int k = 0; k < IN_NODE; ++k) acc += xr[k] * wr[k];
    h[n * HID + t] = (__bf16)acc;
}
__global__ void EdgeGNN_scatter_kernel(const __bf16* __restrict__ h, const int* __restrict__ srcI,
                                       const int* __restrict__ dstI, float* __restrict__ agg) {
    const int e = blockIdx.x;
    const int t = threadIdx.x;                         // 0..127
    atomicAdd(&agg[(size_t)dstI[e] * HID + t], (float)h[(size_t)srcI[e] * HID + t]);
}

// ---------------- SAGEConv layer: h2 = (agg/deg)@Wl^T + bl + h@Wr^T; LN; ReLU ----------------
// 256 thr = 8 waves per 16-node tile. A tiles cooperatively staged to LDS (agg scaled+cvt once);
// Wl then Wr staged to LDS by the Tensor Data Mover, overlapped with A staging.
__global__ void EdgeGNN_layer_kernel(const float* __restrict__ agg, const float* __restrict__ invdeg,
                                     const __bf16* __restrict__ h,
                                     const __bf16* __restrict__ WlB, const float* __restrict__ bl,
                                     const __bf16* __restrict__ WrB,
                                     const float* __restrict__ g,  const float* __restrict__ be,
                                     __bf16* __restrict__ hout) {
    __shared__ __bf16 wsh[HID * HID];                  // 32 KB staged weight (Wl then Wr)
    __shared__ __bf16 aggsh[16][HID];                  // 4 KB  scaled agg tile (bf16)
    __shared__ __bf16 hsh[16][HID];                    // 4 KB  h tile (bf16)
    __shared__ float  tile[16][HID + 4];               // 8.4 KB pre-LN tile
    const int m0   = blockIdx.x * 16;
    const int wave = threadIdx.x >> 5;
    const int lane = threadIdx.x & 31;
    const int row  = lane & 15;
    const int kg   = lane >> 4;
    const int ncol = wave * 16 + row;                  // output column this lane owns in B/D

    // ---- kick off TDM for Wl; cooperatively stage A tiles meanwhile ----
    if (wave == 0) tdm_load_bf16_1d(lds_offset_of(wsh), WlB, HID * HID);
    {
        const int r  = threadIdx.x >> 4;               // 16 rows, 16 threads/row
        const int c0 = (threadIdx.x & 15) * 8;         // 8 elements per thread
        const float  sc = invdeg[m0 + r];
        const float* ap = agg + (size_t)(m0 + r) * HID + c0;
        v8bf o;
#pragma unroll
        for (int j = 0; j < 8; ++j) o[j] = (__bf16)(ap[j] * sc);
        *(v8bf*)&aggsh[r][c0] = o;
        *(v8bf*)&hsh[r][c0]   = *(const v8bf*)(h + (size_t)(m0 + r) * HID + c0);
    }
    if (wave == 0) __builtin_amdgcn_s_wait_tensorcnt(0);
    __syncthreads();

    // ---- GEMM1: (agg/deg) @ Wl^T, both operands in LDS ----
    v8f acc = {};
    const __bf16* w_lds = wsh + (size_t)ncol * HID;
    const __bf16* arow1 = &aggsh[row][0];
#pragma unroll
    for (int kb = 0; kb < HID; kb += 32) {
        v16bf a = loadA_bf16(arow1, kb, kg);
        v16bf b = loadB_bf16(w_lds, kb, kg);
        acc = __builtin_amdgcn_wmma_f32_16x16x32_bf16(false, a, false, b, (short)0, acc, false, false);
    }
    __syncthreads();

    // ---- stage Wr via TDM into the same LDS slot, then GEMM2 (accumulates) ----
    if (wave == 0) {
        tdm_load_bf16_1d(lds_offset_of(wsh), WrB, HID * HID);
        __builtin_amdgcn_s_wait_tensorcnt(0);
    }
    __syncthreads();
    const __bf16* arow2 = &hsh[row][0];
#pragma unroll
    for (int kb = 0; kb < HID; kb += 32) {
        v16bf a = loadA_bf16(arow2, kb, kg);
        v16bf b = loadB_bf16(w_lds, kb, kg);           // now holds Wr
        acc = __builtin_amdgcn_wmma_f32_16x16x32_bf16(false, a, false, b, (short)0, acc, false, false);
    }

    const float bias = bl[ncol];
#pragma unroll
    for (int r = 0; r < 8; ++r)                        // D elem r -> row r + 8*kg, col ncol
        tile[r + kg * 8][ncol] = acc[r] + bias;
    __syncthreads();

    if (threadIdx.x < 16) {                            // LayerNorm + ReLU, one thread per node
        const int rr = threadIdx.x;
        float mu = 0.f;
#pragma unroll 8
        for (int c = 0; c < HID; ++c) mu += tile[rr][c];
        mu *= (1.0f / HID);
        float var = 0.f;
#pragma unroll 8
        for (int c = 0; c < HID; ++c) { float d = tile[rr][c] - mu; var += d * d; }
        var *= (1.0f / HID);
        const float rs = rsqrtf(var + LN_EPS);
        __bf16* orow = hout + (size_t)(m0 + rr) * HID;
#pragma unroll 8
        for (int c = 0; c < HID; ++c) {
            float v = (tile[rr][c] - mu) * rs * g[c] + be[c];
            orow[c] = (__bf16)(v > 0.f ? v : 0.f);
        }
    }
}

// ---------------- fused edge head: concat(h[src],h[dst],ea) -> 128 -> 64 -> 1 ----------------
// 128 thr = 4 waves, 16 edges per wave. K-outer loop: one A fragment feeds 8 (resp. 4)
// independent accumulators. W2 staged to LDS by TDM asynchronously during layer 1.
__global__ void EdgeGNN_edge_head_kernel(const __bf16* __restrict__ h,
                                         const int* __restrict__ srcI, const int* __restrict__ dstI,
                                         const float* __restrict__ ea,
                                         const __bf16* __restrict__ W1p, const float* __restrict__ b1,
                                         const __bf16* __restrict__ W2B, const float* __restrict__ b2,
                                         const float* __restrict__ W3,  const float* __restrict__ b3,
                                         float* __restrict__ out) {
    __shared__ __bf16 e1[4][16][HID];                  // 16 KB
    __shared__ float  e2[4][16][64];                   // 16 KB
    __shared__ __bf16 w2sh[64 * HID];                  // 16 KB
    const int wave = threadIdx.x >> 5;
    const int lane = threadIdx.x & 31;
    const int row  = lane & 15;
    const int kg   = lane >> 4;
    const int e0   = (blockIdx.x * 4 + wave) * 16;
    const int myE  = e0 + row;

    if (wave == 0) tdm_load_bf16_1d(lds_offset_of(w2sh), W2B, 64 * HID);   // async W2 staging

    const __bf16* hs  = h + (size_t)srcI[myE] * HID;
    const __bf16* hd  = h + (size_t)dstI[myE] * HID;
    const float*  eaR = ea + (size_t)myE * IN_EDGE;

    // 8-element bf16 chunk of the virtual row [h_src | h_dst | ea | 0]; k8 is a multiple of 8
    auto chunk8 = [&](int k8) -> v8bf {
        if (k8 < 128) return *(const v8bf*)(hs + k8);
        if (k8 < 256) return *(const v8bf*)(hd + (k8 - 128));
        v8bf t;
        if (k8 == 256) {
#pragma unroll
            for (int j = 0; j < 8; ++j) t[j] = (__bf16)eaR[j];
        } else {
#pragma unroll
            for (int j = 0; j < 8; ++j) t[j] = (__bf16)0.f;
        }
        return t;
    };
    const int nB = lane & 15;

    // ---- layer 1: [16,288] x [288,128]; K outer, 8 accumulators ----
    {
        v8f acc[8];
#pragma unroll
        for (int nt = 0; nt < 8; ++nt) acc[nt] = (v8f){};
        const __bf16* w1base = W1p + (size_t)nB * K1PAD;
#pragma unroll
        for (int kb = 0; kb < K1PAD; kb += 32) {
            v8bf lo = chunk8(kb + kg * 8);
            v8bf hi = chunk8(kb + 16 + kg * 8);
            v16bf a;
#pragma unroll
            for (int j = 0; j < 8; ++j) { a[j] = lo[j]; a[8 + j] = hi[j]; }
#pragma unroll
            for (int nt = 0; nt < 8; ++nt) {
                v16bf b = loadB_bf16(w1base + (size_t)nt * 16 * K1PAD, kb, kg);
                acc[nt] = __builtin_amdgcn_wmma_f32_16x16x32_bf16(false, a, false, b, (short)0, acc[nt], false, false);
            }
        }
#pragma unroll
        for (int nt = 0; nt < 8; ++nt) {
            const float bias = b1[nt * 16 + nB];
#pragma unroll
            for (int r = 0; r < 8; ++r) {
                float v = acc[nt][r] + bias;
                e1[wave][r + kg * 8][nt * 16 + nB] = (__bf16)(v > 0.f ? v : 0.f);
            }
        }
    }
    __syncthreads();
    if (wave == 0) __builtin_amdgcn_s_wait_tensorcnt(0);   // W2 now resident in LDS
    __syncthreads();

    // ---- layer 2: [16,128] x [128,64]; K outer, 4 accumulators, all operands in LDS ----
    {
        v8f acc[4];
#pragma unroll
        for (int nt = 0; nt < 4; ++nt) acc[nt] = (v8f){};
        const __bf16* arow   = &e1[wave][row][0];
        const __bf16* w2base = w2sh + (size_t)nB * HID;
#pragma unroll
        for (int kb = 0; kb < HID; kb += 32) {
            v16bf a = loadA_bf16(arow, kb, kg);
#pragma unroll
            for (int nt = 0; nt < 4; ++nt) {
                v16bf b = loadB_bf16(w2base + (size_t)nt * 16 * HID, kb, kg);
                acc[nt] = __builtin_amdgcn_wmma_f32_16x16x32_bf16(false, a, false, b, (short)0, acc[nt], false, false);
            }
        }
#pragma unroll
        for (int nt = 0; nt < 4; ++nt) {
            const float bias = b2[nt * 16 + nB];
#pragma unroll
            for (int r = 0; r < 8; ++r) {
                float v = acc[nt][r] + bias;
                e2[wave][r + kg * 8][nt * 16 + nB] = v > 0.f ? v : 0.f;
            }
        }
    }
    __syncthreads();

    // ---- layer 3: per-edge dot(64) ----
    if (lane < 16) {
        float acc3 = b3[0];
#pragma unroll
        for (int c = 0; c < 64; ++c) acc3 += e2[wave][lane][c] * W3[c];
        out[e0 + lane] = acc3;
    }
}

extern "C" void kernel_launch(void* const* d_in, const int* in_sizes, int n_in,
                              void* d_out, int out_size, void* d_ws, size_t ws_size,
                              hipStream_t stream) {
    (void)in_sizes; (void)n_in; (void)out_size; (void)ws_size;
    const float* x   = (const float*)d_in[0];
    const int*   ei  = (const int*)  d_in[1];
    const float* ea  = (const float*)d_in[2];
    const float* Win = (const float*)d_in[3];
    const float* bin = (const float*)d_in[4];
    const float* Wl  = (const float*)d_in[5];
    const float* bl  = (const float*)d_in[6];
    const float* Wr  = (const float*)d_in[7];
    const float* lng = (const float*)d_in[8];
    const float* lnb = (const float*)d_in[9];
    const float* W1  = (const float*)d_in[10];
    const float* b1  = (const float*)d_in[11];
    const float* W2  = (const float*)d_in[12];
    const float* b2  = (const float*)d_in[13];
    const float* W3  = (const float*)d_in[14];
    const float* b3  = (const float*)d_in[15];
    float* out = (float*)d_out;
    const int* srcI = ei;
    const int* dstI = ei + N_EDGES;

    // workspace layout (256-aligned)
    char* ws = (char*)d_ws;
    const size_t AGG_B = (size_t)N_NODES * HID * sizeof(float);   // 25,600,000
    const size_t HB_B  = (size_t)N_NODES * HID * sizeof(__bf16);  // 12,800,000
    float*  agg  = (float*)(ws);
    __bf16* hA   = (__bf16*)(ws + AGG_B);
    __bf16* hB   = (__bf16*)(ws + AGG_B + HB_B);
    float*  deg  = (float*)(ws + AGG_B + 2 * HB_B);               // 200,000 B
    char*   wtop = ws + AGG_B + 2 * HB_B + 200192;
    __bf16* WlB  = (__bf16*)(wtop);                               // 4*128*128*2 = 131072
    __bf16* WrB  = (__bf16*)(wtop + 131072);                      // 131072
    __bf16* W1p  = (__bf16*)(wtop + 262144);                      // 128*288*2 = 73728
    __bf16* W2B  = (__bf16*)(wtop + 335872);                      // 64*128*2 = 16384

    // weight conversions (once per launch)
    EdgeGNN_cvt_kernel<<<(N_LAYERS * HID * HID + 255) / 256, 256, 0, stream>>>(Wl, WlB, N_LAYERS * HID * HID);
    EdgeGNN_cvt_kernel<<<(N_LAYERS * HID * HID + 255) / 256, 256, 0, stream>>>(Wr, WrB, N_LAYERS * HID * HID);
    EdgeGNN_cvtpad_w1_kernel<<<(HID * K1PAD + 255) / 256, 256, 0, stream>>>(W1, W1p);
    EdgeGNN_cvt_kernel<<<(64 * HID + 255) / 256, 256, 0, stream>>>(W2, W2B, 64 * HID);

    hipMemsetAsync(deg, 0, (size_t)N_NODES * sizeof(float), stream);
    EdgeGNN_deg_kernel<<<(N_EDGES + 255) / 256, 256, 0, stream>>>(dstI, deg);
    EdgeGNN_invdeg_kernel<<<(N_NODES + 255) / 256, 256, 0, stream>>>(deg);
    EdgeGNN_inproj_kernel<<<N_NODES, HID, 0, stream>>>(x, Win, bin, hA);

    __bf16* hin = hA;
    __bf16* hout = hB;
    for (int l = 0; l < N_LAYERS; ++l) {
        hipMemsetAsync(agg, 0, AGG_B, stream);
        EdgeGNN_scatter_kernel<<<N_EDGES, HID, 0, stream>>>(hin, srcI, dstI, agg);
        EdgeGNN_layer_kernel<<<N_NODES / 16, 256, 0, stream>>>(
            agg, deg, hin,
            WlB + (size_t)l * HID * HID, bl + (size_t)l * HID,
            WrB + (size_t)l * HID * HID,
            lng + (size_t)l * HID, lnb + (size_t)l * HID, hout);
        __bf16* t = hin; hin = hout; hout = t;
    }

    EdgeGNN_edge_head_kernel<<<N_EDGES / 64, 128, 0, stream>>>(
        hin, srcI, dstI, ea, W1p, b1, W2B, b2, W3, b3, out);
}
